// DariushFlashAttention2_14087492731054
// MI455X (gfx1250) — matrix operand order repaired and checked by reading, not compiled, not run on hardware
//
#include <hip/hip_runtime.h>

typedef _Float16 v16h __attribute__((ext_vector_type(16)));
typedef _Float16 v8h  __attribute__((ext_vector_type(8)));
typedef float    v8f  __attribute__((ext_vector_type(8)));
typedef float    v4f  __attribute__((ext_vector_type(4)));

#define HEADS 32
#define DIM   128
#define TOK_ELEMS (HEADS * DIM)   // 4096 floats per token per tensor
#define QK_PITCH 136              // halves per row: 272B = 68 dwords (== 4 mod 64 banks)
#define VT_PITCH 40               // halves per row: 80B  = 20 dwords (spreads banks)

// One workgroup (4 wave32) per token. Per-token head-attention:
//   S = (Q * 1/sqrt(D)) K^T   (32x32, v_wmma_f32_16x16x32_f16, 2x2 tiles x 4 k-steps)
//   P = softmax_rows(S)       (exact f32 softmax)
//   O = P V                   (32x128, 2x8 tiles, 1 k-step each)
__launch_bounds__(128)
__global__ void head_attn_wmma_kernel(const float* __restrict__ q,
                                      const float* __restrict__ k,
                                      const float* __restrict__ v,
                                      float* __restrict__ out)
{
    __shared__ __attribute__((aligned(32))) _Float16 Qh[HEADS * QK_PITCH]; // [h][d] pre-scaled
    __shared__ __attribute__((aligned(32))) _Float16 Kh[HEADS * QK_PITCH]; // [g][d]
    __shared__ __attribute__((aligned(32))) _Float16 Vt[DIM * VT_PITCH];   // [d][g] transposed
    __shared__ __attribute__((aligned(32))) float    Sf[HEADS * HEADS];
    __shared__ __attribute__((aligned(32))) _Float16 Ph[HEADS * HEADS];
    __shared__ __attribute__((aligned(16))) float    Of[TOK_ELEMS];        // output staging

    const int tid  = threadIdx.x;          // 0..127
    const int wave = tid >> 5;             // 0..3
    const int lane = tid & 31;
    const int l15  = lane & 15;
    const int hs   = lane >> 4;            // half-wave select (0 | 1)

    const size_t base  = (size_t)blockIdx.x * TOK_ELEMS;
    const float  scale = 0.08838834764831845f;   // 1/sqrt(128)

    // ---- Stage 1a: Q/K coalesced float4 loads -> f16 LDS rows (padded pitch) ----
    const v4f* q4 = (const v4f*)(q + base);
    const v4f* k4 = (const v4f*)(k + base);
    #pragma unroll
    for (int i = 0; i < 8; ++i) {
        const int idx = tid + i * 128;     // float4 index, 0..1023
        const v4f qv = __builtin_nontemporal_load(&q4[idx]);
        const v4f kv = __builtin_nontemporal_load(&k4[idx]);
        const int e = idx * 4;
        const int h = e >> 7;              // head row
        const int d = e & 127;             // dim within row (4-aligned)
        _Float16* qd = &Qh[h * QK_PITCH + d];
        _Float16* kd = &Kh[h * QK_PITCH + d];
        #pragma unroll
        for (int j = 0; j < 4; ++j) {
            qd[j] = (_Float16)(qv[j] * scale);
            kd[j] = (_Float16)kv[j];
        }
    }

    // ---- Stage 1b: V -> V^T. Thread tid owns column d = tid; loads stay coalesced
    // (iteration g reads 128 consecutive floats), stores are contiguous b128 rows. ----
    {
        const float* vg = v + base + tid;
        v8h vc[4];
        #pragma unroll
        for (int g = 0; g < HEADS; ++g)
            vc[g >> 3][g & 7] = (_Float16)__builtin_nontemporal_load(vg + g * DIM);
        #pragma unroll
        for (int c = 0; c < 4; ++c)
            *(v8h*)(&Vt[tid * VT_PITCH + c * 8]) = vc[c];
    }
    __syncthreads();

    // ---- Stage 2: S = Q K^T (scaled), one 16x16 C-tile per wave ----
    {
        const int mi = wave >> 1;          // M tile (q-head block)
        const int ni = wave & 1;           // N tile (k-head block)
        const int hrow = mi * 16 + l15;    // A row  = q head
        const int gcol = ni * 16 + l15;    // B col  = k head
        v8f acc = {};
        #pragma unroll
        for (int ks = 0; ks < 4; ++ks) {
            // A fragment: lanes 0-15 -> K {0..15}; lanes 16-31 -> {16..31}
            const _Float16* ab = &Qh[hrow * QK_PITCH + ks * 32 + hs * 16];
            const v8h alo = *(const v8h*)(ab);
            const v8h ahi = *(const v8h*)(ab + 8);
            const v16h a = __builtin_shufflevector(alo, ahi,
                0, 1, 2, 3, 4, 5, 6, 7, 8, 9, 10, 11, 12, 13, 14, 15);
            // B fragment: lane = column g, 16 contiguous K (d) values
            const v16h b = *(const v16h*)(&Kh[gcol * QK_PITCH + ks * 32 + hs * 16]);
            acc = __builtin_amdgcn_wmma_f32_16x16x32_f16(
                false, a, false, b, (short)0, acc, false, false);
        }
        // C layout: VGPR r = row (mi*16 + hs*8 + r), lane = col
        #pragma unroll
        for (int r = 0; r < 8; ++r)
            Sf[(mi * 16 + hs * 8 + r) * HEADS + gcol] = acc[r];
    }
    __syncthreads();

    // ---- Stage 3: exact f32 row-softmax over the head axis ----
    if (tid < HEADS) {
        float row[HEADS];
        float m = -INFINITY;
        #pragma unroll
        for (int g = 0; g < HEADS; ++g) {
            row[g] = Sf[tid * HEADS + g];
            m = fmaxf(m, row[g]);
        }
        float s = 0.0f;
        #pragma unroll
        for (int g = 0; g < HEADS; ++g) {
            row[g] = __expf(row[g] - m);
            s += row[g];
        }
        const float inv = 1.0f / s;
        #pragma unroll
        for (int g = 0; g < HEADS; ++g)
            Ph[tid * HEADS + g] = (_Float16)(row[g] * inv);
    }
    __syncthreads();

    // ---- Stage 4: O = P V, 4 tiles per wave, single k-step each ----
    {
        const int mi = wave & 1;
        const int hrow = mi * 16 + l15;
        const _Float16* pab = &Ph[hrow * HEADS + hs * 16];
        const v8h plo = *(const v8h*)(pab);
        const v8h phi = *(const v8h*)(pab + 8);
        const v16h a = __builtin_shufflevector(plo, phi,
            0, 1, 2, 3, 4, 5, 6, 7, 8, 9, 10, 11, 12, 13, 14, 15);
        #pragma unroll
        for (int t = 0; t < 4; ++t) {
            const int ni = (wave >> 1) + t * 2;      // all 8 N tiles covered across waves
            const int dcol = ni * 16 + l15;          // output dim column
            // B fragment from V^T: lane = column d, 16 contiguous g values
            const v16h b = *(const v16h*)(&Vt[dcol * VT_PITCH + hs * 16]);
            v8f acc = {};
            acc = __builtin_amdgcn_wmma_f32_16x16x32_f16(
                false, a, false, b, (short)0, acc, false, false);
            #pragma unroll
            for (int r = 0; r < 8; ++r)
                Of[(mi * 16 + hs * 8 + r) * DIM + dcol] = acc[r];
        }
    }
    __syncthreads();

    // ---- Stage 5: coalesced non-temporal float4 store of the token's output ----
    v4f* o4 = (v4f*)(out + base);
    const v4f* of4 = (const v4f*)Of;
    #pragma unroll
    for (int i = 0; i < 8; ++i)
        __builtin_nontemporal_store(of4[tid + i * 128], &o4[tid + i * 128]);
}

extern "C" void kernel_launch(void* const* d_in, const int* in_sizes, int n_in,
                              void* d_out, int out_size, void* d_ws, size_t ws_size,
                              hipStream_t stream) {
    (void)n_in; (void)out_size; (void)d_ws; (void)ws_size;
    const float* q = (const float*)d_in[0];
    const float* k = (const float*)d_in[1];
    const float* v = (const float*)d_in[2];
    float* out = (float*)d_out;
    const int tokens = in_sizes[0] / TOK_ELEMS;   // B * S = 8192
    head_attn_wmma_kernel<<<tokens, 128, 0, stream>>>(q, k, v, out);
}
